// MambaCAN_noid_6803228196994
// MI455X (gfx1250) — matrix operand order repaired and checked by reading, not compiled, not run on hardware
//
#include <hip/hip_runtime.h>

typedef __bf16 bf16;
typedef __attribute__((ext_vector_type(16))) __bf16 v16bf;
typedef __attribute__((ext_vector_type(8)))  __bf16 v8bf;
typedef __attribute__((ext_vector_type(8)))  float  v8f;

#define BSZ  4
#define SEQ  2048
#define DM   256
#define DI   512
#define DSTATE 16
#define HIDC 128
#define LP   (SEQ + 2)   // conv2 input rows with 1-row zero halo each side

// ---------------------------------------------------------------------------
// WMMA fragment helpers (bf16 16x16x32, wave32). Layouts per CDNA5 ISA.
// A-matrix 16x32: lane(0-15)=row M, kh=lane>>4 selects K {0-7,16-23} vs {8-15,24-31}.
__device__ __forceinline__ v16bf load_fragA(const bf16* __restrict__ base, int ld) {
    const int lane = threadIdx.x & 31;
    const int row  = lane & 15;
    const int kh   = lane >> 4;
    const bf16* p  = base + (size_t)row * ld + kh * 8;
    v8bf lo = *reinterpret_cast<const v8bf*>(p);
    v8bf hi = *reinterpret_cast<const v8bf*>(p + 16);
    v16bf f;
#pragma unroll
    for (int i = 0; i < 8; ++i) { f[i] = lo[i]; f[i + 8] = hi[i]; }
    return f;
}

// B-matrix 32x16 taken as W.T where W is row-major (N x K):
// lane(0-15)=col N, kh selects K {0-7,16-23} vs {8-15,24-31} (same as A).
__device__ __forceinline__ v16bf load_fragB(const bf16* __restrict__ Wbase, int ldw) {
    const int lane = threadIdx.x & 31;
    const int col  = lane & 15;
    const int kh   = lane >> 4;
    const bf16* p  = Wbase + (size_t)col * ldw + kh * 8;
    v8bf lo = *reinterpret_cast<const v8bf*>(p);
    v8bf hi = *reinterpret_cast<const v8bf*>(p + 16);
    v16bf f;
#pragma unroll
    for (int i = 0; i < 8; ++i) { f[i] = lo[i]; f[i + 8] = hi[i]; }
    return f;
}

__device__ __forceinline__ v8f wmma_bf16(v16bf a, v16bf b, v8f c) {
    return __builtin_amdgcn_wmma_f32_16x16x32_bf16(false, a, false, b, (short)0, c,
                                                   false, false);
}

// ---------------------------------------------------------------------------
__global__ void k_f32_to_bf16(const float* __restrict__ src, bf16* __restrict__ dst, int n) {
    int i = blockIdx.x * blockDim.x + threadIdx.x;
    if (i < n) dst[i] = (bf16)src[i];
}

__global__ void k_zero_bf16(bf16* __restrict__ dst, int n) {
    int i = blockIdx.x * blockDim.x + threadIdx.x;
    if (i < n) dst[i] = (bf16)0.0f;
}

// conv2 weight (O=256, C=128, K=3) -> bf16 (K, O, C) contiguous-in-C
__global__ void k_pack_w2(const float* __restrict__ src, bf16* __restrict__ dst) {
    int i = blockIdx.x * blockDim.x + threadIdx.x;
    if (i >= 3 * 256 * 128) return;
    int c = i % 128;
    int o = (i / 128) % 256;
    int k = i / (128 * 256);
    dst[i] = (bf16)src[o * 384 + c * 3 + k];
}

// conv1 (9 -> 128, k=3, pad 1) + ReLU, write bf16 into halo-padded f1p
__global__ void k_conv1(const float* __restrict__ x, const float* __restrict__ w,
                        const float* __restrict__ b, bf16* __restrict__ f1p) {
    int l  = blockIdx.x;
    int bb = blockIdx.y;
    int o  = threadIdx.x;
    float acc = b[o];
#pragma unroll
    for (int k = 0; k < 3; ++k) {
        int li = l + k - 1;
        if (li < 0 || li >= SEQ) continue;
        const float* xr = x + ((size_t)bb * SEQ + li) * 9;
#pragma unroll
        for (int i = 0; i < 9; ++i) acc += xr[i] * w[o * 27 + i * 3 + k];
    }
    f1p[((size_t)bb * LP + l + 1) * HIDC + o] = (bf16)fmaxf(acc, 0.0f);
}

// conv2 (128 -> 256, k=3, pad 1) as 3 shift-GEMMs over padded f1p.
// One wave computes a 32x32 C tile (2x2 WMMA register tiling); + bias + ReLU -> bf16 feat
__global__ void k_conv2_wmma(const bf16* __restrict__ f1p, const bf16* __restrict__ w2p,
                             const float* __restrict__ b2, bf16* __restrict__ feat) {
    int wid  = blockIdx.x * (blockDim.x >> 5) + (threadIdx.x >> 5);
    int nt2  = wid & 7;           // 8 tiles of N=256 (32 wide)
    int rest = wid >> 3;
    int lt2  = rest & 63;         // 64 tiles of L=2048 (32 tall)
    int bb   = rest >> 6;
    if (bb >= BSZ) return;
    v8f acc00 = {}, acc01 = {}, acc10 = {}, acc11 = {};
#pragma unroll
    for (int k = 0; k < 3; ++k) {
        const bf16* a0 = f1p + ((size_t)bb * LP + lt2 * 32 + k) * HIDC;
        const bf16* a1 = a0 + (size_t)16 * HIDC;
        const bf16* w0 = w2p + (size_t)k * 256 * HIDC + (size_t)nt2 * 32 * HIDC;
        const bf16* w1 = w0 + (size_t)16 * HIDC;
#pragma unroll
        for (int kk = 0; kk < HIDC; kk += 32) {
            v16bf A0 = load_fragA(a0 + kk, HIDC);
            v16bf A1 = load_fragA(a1 + kk, HIDC);
            v16bf B0 = load_fragB(w0 + kk, HIDC);
            v16bf B1 = load_fragB(w1 + kk, HIDC);
            acc00 = wmma_bf16(A0, B0, acc00);
            acc01 = wmma_bf16(A0, B1, acc01);
            acc10 = wmma_bf16(A1, B0, acc10);
            acc11 = wmma_bf16(A1, B1, acc11);
        }
    }
    int lane = threadIdx.x & 31, col = lane & 15, kh = lane >> 4;
    float bias0 = b2[nt2 * 32 + col];
    float bias1 = b2[nt2 * 32 + 16 + col];
    bf16* out = feat + ((size_t)bb * SEQ + lt2 * 32) * DM + nt2 * 32;
#pragma unroll
    for (int r = 0; r < 8; ++r) {
        size_t r0 = (size_t)(r + 8 * kh) * DM;
        size_t r1 = (size_t)(r + 8 * kh + 16) * DM;
        out[r0 + col]      = (bf16)fmaxf(acc00[r] + bias0, 0.0f);
        out[r0 + 16 + col] = (bf16)fmaxf(acc01[r] + bias1, 0.0f);
        out[r1 + col]      = (bf16)fmaxf(acc10[r] + bias0, 0.0f);
        out[r1 + 16 + col] = (bf16)fmaxf(acc11[r] + bias1, 0.0f);
    }
}

// Generic C = A(M x K, bf16) @ W(N x K, bf16).T -> f32.
// 2x2 register tiling: one wave per 32x32 C tile (M, N multiples of 32).
__global__ void k_gemm_wmma2(const bf16* __restrict__ A, const bf16* __restrict__ W,
                             float* __restrict__ C, int Mt2, int Nt2, int K) {
    int wid = blockIdx.x * (blockDim.x >> 5) + (threadIdx.x >> 5);
    int nt  = wid % Nt2;
    int mt  = wid / Nt2;
    if (mt >= Mt2) return;
    const int N = Nt2 * 32;
    v8f acc00 = {}, acc01 = {}, acc10 = {}, acc11 = {};
    const bf16* a0 = A + (size_t)mt * 32 * K;
    const bf16* a1 = a0 + (size_t)16 * K;
    const bf16* w0 = W + (size_t)nt * 32 * K;
    const bf16* w1 = w0 + (size_t)16 * K;
    for (int kk = 0; kk < K; kk += 32) {
        v16bf A0 = load_fragA(a0 + kk, K);
        v16bf A1 = load_fragA(a1 + kk, K);
        v16bf B0 = load_fragB(w0 + kk, K);
        v16bf B1 = load_fragB(w1 + kk, K);
        acc00 = wmma_bf16(A0, B0, acc00);
        acc01 = wmma_bf16(A0, B1, acc01);
        acc10 = wmma_bf16(A1, B0, acc10);
        acc11 = wmma_bf16(A1, B1, acc11);
    }
    int lane = threadIdx.x & 31, col = lane & 15, kh = lane >> 4;
    float* out = C + (size_t)mt * 32 * N + nt * 32;
#pragma unroll
    for (int r = 0; r < 8; ++r) {
        size_t r0 = (size_t)(r + 8 * kh) * N;
        size_t r1 = (size_t)(r + 8 * kh + 16) * N;
        out[r0 + col]      = acc00[r];
        out[r0 + 16 + col] = acc01[r];
        out[r1 + col]      = acc10[r];
        out[r1 + 16 + col] = acc11[r];
    }
}

// Generic 16x16-tile GEMM (for N not a multiple of 32, e.g. x_proj N=48)
__global__ void k_gemm_wmma(const bf16* __restrict__ A, const bf16* __restrict__ W,
                            float* __restrict__ C, int Mt, int Nt, int K) {
    int wid = blockIdx.x * (blockDim.x >> 5) + (threadIdx.x >> 5);
    int nt  = wid % Nt;
    int mt  = wid / Nt;
    if (mt >= Mt) return;
    const int N = Nt * 16;
    v8f acc = {};
    const bf16* abase = A + (size_t)mt * 16 * K;
    const bf16* wbase = W + (size_t)nt * 16 * K;
    for (int kk = 0; kk < K; kk += 32) {
        v16bf af = load_fragA(abase + kk, K);
        v16bf bf = load_fragB(wbase + kk, K);
        acc = wmma_bf16(af, bf, acc);
    }
    int lane = threadIdx.x & 31, col = lane & 15, kh = lane >> 4;
    float* out = C + (size_t)mt * 16 * N + nt * 16;
#pragma unroll
    for (int r = 0; r < 8; ++r) out[(size_t)(r + 8 * kh) * N + col] = acc[r];
}

// depthwise causal conv (k=4, pad left 3) + SiLU; emits u in f32 and bf16
__global__ void k_dwconv_silu(const float* __restrict__ xz, const float* __restrict__ w,
                              const float* __restrict__ bias, float* __restrict__ uf,
                              bf16* __restrict__ ub) {
    int idx = blockIdx.x * blockDim.x + threadIdx.x;
    if (idx >= BSZ * SEQ * DI) return;
    int d  = idx % DI;
    int t  = (idx / DI) % SEQ;
    int bb = idx / (DI * SEQ);
    float acc = bias[d];
#pragma unroll
    for (int k = 0; k < 4; ++k) {
        int ti = t - 3 + k;
        if (ti >= 0) acc += xz[((size_t)bb * SEQ + ti) * 1024 + d] * w[d * 4 + k];
    }
    float s = acc / (1.0f + __expf(-acc));
    uf[idx] = s;
    ub[idx] = (bf16)s;
}

// dt = softplus(dt_lr @ dtW.T + dtB)  (K=16, scalar dot is cheap)
__global__ void k_dtproj(const float* __restrict__ xdbl, const float* __restrict__ w,
                         const float* __restrict__ b, float* __restrict__ dt) {
    int idx = blockIdx.x * blockDim.x + threadIdx.x;
    if (idx >= BSZ * SEQ * DI) return;
    int d   = idx % DI;
    int row = idx / DI;
    const float* xr = xdbl + (size_t)row * 48;
    float acc = b[d];
#pragma unroll
    for (int r = 0; r < 16; ++r) acc += xr[r] * w[d * 16 + r];
    dt[idx] = (acc > 20.0f) ? acc : log1pf(__expf(acc));
}

// selective scan: one lane per (b, d); 16-wide state in registers; B/C staged in LDS
// output fused with +u*D, gate silu(z), written bf16 for out_proj WMMA
__global__ void k_scan(const float* __restrict__ uf, const float* __restrict__ dt,
                       const float* __restrict__ xdbl, const float* __restrict__ xz,
                       const float* __restrict__ A_log, const float* __restrict__ Dp,
                       bf16* __restrict__ yg) {
    __shared__ float sB[64][DSTATE];
    __shared__ float sC[64][DSTATE];
    int bb = blockIdx.x >> 1;
    int d  = ((blockIdx.x & 1) << 8) + threadIdx.x;
    float a[DSTATE], h[DSTATE];
#pragma unroll
    for (int s = 0; s < DSTATE; ++s) {
        a[s] = -__expf(A_log[d * DSTATE + s]);
        h[s] = 0.0f;
    }
    float Dd = Dp[d];
    const size_t rowbase = (size_t)bb * SEQ;
    for (int tc = 0; tc < SEQ / 64; ++tc) {
#pragma unroll
        for (int j = 0; j < 4; ++j) {
            int e = j * 256 + threadIdx.x;
            int tt = e >> 4, s = e & 15;
            const float* xr = xdbl + (rowbase + tc * 64 + tt) * 48;
            sB[tt][s] = xr[16 + s];
            sC[tt][s] = xr[32 + s];
        }
        __syncthreads();
        for (int t = 0; t < 64; ++t) {
            size_t row = rowbase + tc * 64 + t;
            float dtv = dt[row * DI + d];
            float uv  = uf[row * DI + d];
            float dtu = dtv * uv;
            float acc = 0.0f;
#pragma unroll
            for (int s = 0; s < DSTATE; ++s) {
                float dA = __expf(dtv * a[s]);
                h[s] = dA * h[s] + dtu * sB[t][s];
                acc += h[s] * sC[t][s];
            }
            float zv = xz[row * 1024 + DI + d];
            float sz = zv / (1.0f + __expf(-zv));
            yg[row * DI + d] = (bf16)((acc + uv * Dd) * sz);
        }
        __syncthreads();
    }
}

__global__ void k_pool(const float* __restrict__ mo, float* __restrict__ pooled) {
    int bb = blockIdx.x;
    int o  = threadIdx.x;
    float m = -3.4e38f;
    for (int t = 0; t < SEQ; ++t) m = fmaxf(m, mo[((size_t)bb * SEQ + t) * DM + o]);
    pooled[bb * DM + o] = m;
}

__global__ void k_head(const float* __restrict__ pooled, const float* __restrict__ w1,
                       const float* __restrict__ b1, const float* __restrict__ w2,
                       const float* __restrict__ b2, float* __restrict__ out) {
    __shared__ float sh[BSZ * HIDC];
    int j = threadIdx.x;
    for (int bb = 0; bb < BSZ; ++bb) {
        float acc = b1[j];
        for (int i = 0; i < DM; ++i) acc += pooled[bb * DM + i] * w1[j * DM + i];
        sh[bb * HIDC + j] = fmaxf(acc, 0.0f);
    }
    __syncthreads();
    if (j < BSZ * 4) {
        int bb = j >> 2, c = j & 3;
        float acc = b2[c];
        for (int i = 0; i < HIDC; ++i) acc += sh[bb * HIDC + i] * w2[c * HIDC + i];
        out[bb * 4 + c] = acc;
    }
}

// ---------------------------------------------------------------------------
extern "C" void kernel_launch(void* const* d_in, const int* in_sizes, int n_in,
                              void* d_out, int out_size, void* d_ws, size_t ws_size,
                              hipStream_t stream) {
    (void)in_sizes; (void)n_in; (void)out_size; (void)ws_size;
    const float* x    = (const float*)d_in[0];
    const float* w1   = (const float*)d_in[1];
    const float* b1   = (const float*)d_in[2];
    const float* w2   = (const float*)d_in[3];
    const float* b2   = (const float*)d_in[4];
    const float* win  = (const float*)d_in[5];
    const float* wdw  = (const float*)d_in[6];
    const float* bdw  = (const float*)d_in[7];
    const float* wxp  = (const float*)d_in[8];
    const float* wdt  = (const float*)d_in[9];
    const float* bdt  = (const float*)d_in[10];
    const float* alog = (const float*)d_in[11];
    const float* Dp   = (const float*)d_in[12];
    const float* wout = (const float*)d_in[13];
    const float* wfc1 = (const float*)d_in[14];
    const float* bfc1 = (const float*)d_in[15];
    const float* wfc2 = (const float*)d_in[16];
    const float* bfc2 = (const float*)d_in[17];
    float* out = (float*)d_out;

    char* base = (char*)d_ws;
    size_t off = 0;
    auto carve = [&](size_t bytes) -> void* {
        void* p = base + off;
        off += (bytes + 255) & ~(size_t)255;
        return p;
    };
    const size_t ROWS = (size_t)BSZ * SEQ;           // 8192
    bf16*  f1p   = (bf16*) carve((size_t)BSZ * LP * HIDC * 2);   // padded conv1 out
    bf16*  feat  = (bf16*) carve(ROWS * DM * 2);                 // conv2 out
    float* xz    = (float*)carve(ROWS * 1024 * 4);               // in_proj out
    float* uf    = (float*)carve(ROWS * DI * 4);                 // silu(dwconv) f32
    bf16*  ub    = (bf16*) carve(ROWS * DI * 2);                 // same, bf16
    float* xdbl  = (float*)carve(ROWS * 48 * 4);                 // x_proj out
    float* dtbuf = (float*)carve(ROWS * DI * 4);                 // softplus dt
    bf16*  yg    = (bf16*) carve(ROWS * DI * 2);                 // gated scan out
    float* mo    = (float*)carve(ROWS * DM * 4);                 // out_proj out
    float* pooled= (float*)carve((size_t)BSZ * DM * 4);
    bf16*  win_b = (bf16*) carve((size_t)1024 * 256 * 2);
    bf16*  wxp_b = (bf16*) carve((size_t)48 * 512 * 2);
    bf16*  wout_b= (bf16*) carve((size_t)256 * 512 * 2);
    bf16*  w2p   = (bf16*) carve((size_t)3 * 256 * 128 * 2);

    // weight conversions (independent)
    k_f32_to_bf16<<<(1024 * 256 + 255) / 256, 256, 0, stream>>>(win, win_b, 1024 * 256);
    k_f32_to_bf16<<<(48 * 512 + 255) / 256, 256, 0, stream>>>(wxp, wxp_b, 48 * 512);
    k_f32_to_bf16<<<(256 * 512 + 255) / 256, 256, 0, stream>>>(wout, wout_b, 256 * 512);
    k_pack_w2<<<(3 * 256 * 128 + 255) / 256, 256, 0, stream>>>(w2, w2p);

    // front convs
    int nf1 = BSZ * LP * HIDC;
    k_zero_bf16<<<(nf1 + 255) / 256, 256, 0, stream>>>(f1p, nf1);
    k_conv1<<<dim3(SEQ, BSZ), HIDC, 0, stream>>>(x, w1, b1, f1p);
    // conv2: BSZ * 64 ltile * 8 ntile = 2048 waves, 8 waves/block
    k_conv2_wmma<<<(BSZ * 64 * 8) / 8, 256, 0, stream>>>(f1p, w2p, b2, feat);

    // in_proj: (8192x256)@(256x1024)^T  -> 256 x 32 tiles of 32x32
    k_gemm_wmma2<<<(256 * 32) / 8, 256, 0, stream>>>(feat, win_b, xz, 256, 32, 256);

    // depthwise conv + silu
    k_dwconv_silu<<<(BSZ * SEQ * DI) / 256, 256, 0, stream>>>(xz, wdw, bdw, uf, ub);

    // x_proj: (8192x512)@(512x48)^T  (N=48 -> 16x16 kernel)
    k_gemm_wmma<<<(512 * 3) / 8, 256, 0, stream>>>(ub, wxp_b, xdbl, 512, 3, 512);

    // dt projection + softplus
    k_dtproj<<<(BSZ * SEQ * DI) / 256, 256, 0, stream>>>(xdbl, wdt, bdt, dtbuf);

    // selective scan (fused +u*D, gate silu(z))
    k_scan<<<BSZ * (DI / 256), 256, 0, stream>>>(uf, dtbuf, xdbl, xz, alog, Dp, yg);

    // out_proj: (8192x512)@(512x256)^T -> 256 x 8 tiles of 32x32
    k_gemm_wmma2<<<(256 * 8) / 8, 256, 0, stream>>>(yg, wout_b, mo, 256, 8, 512);

    // max pool over L, then MLP head
    k_pool<<<BSZ, DM, 0, stream>>>(mo, pooled);
    k_head<<<1, HIDC, 0, stream>>>(pooled, wfc1, bfc1, wfc2, bfc2, out);
}
